// PyGRUCell_89885075571127
// MI455X (gfx1250) — compile-verified, matrix operand order checked by reading, not run on hardware
//
#include <hip/hip_runtime.h>
#include <hip/hip_bf16.h>

// ---------------------------------------------------------------------------
// GRU on MI455X (gfx1250, wave32, WMMA).
// Phase 1: x projections as double-buffered LDS WMMA GEMM, staged with
//          CDNA5 async global->LDS copies (ASYNCcnt path).
// Phase 2: persistent scan kernel, recurrent weights resident in LDS,
//          h state ping-ponged through L2, grid barrier per timestep.
// Workspace requirement: ~228 MB.
// ---------------------------------------------------------------------------

typedef __attribute__((ext_vector_type(16))) __bf16 v16bf;
typedef __attribute__((ext_vector_type(8)))  float  v8f;

static constexpr int B = 32, L = 1024, D = 512, H = 512;
static constexpr int M_TOT = B * L;              // 32768 rows of x

// ---- workspace offsets (bytes) --------------------------------------------
static constexpr size_t OFF_SYNC = 0;                                   // 4KB
static constexpr size_t OFF_HF   = 4096;                                // 2 * 32*512 f32
static constexpr size_t OFF_HB   = OFF_HF  + (size_t)2 * B * H * 4;     // 2 * 32*512 bf16
static constexpr size_t OFF_WXT  = OFF_HB  + (size_t)2 * B * H * 2;     // 3 * 512*512 bf16
static constexpr size_t OFF_WHT  = OFF_WXT + (size_t)3 * D * H * 2;     // 3 * 512*512 bf16
static constexpr size_t OFF_XB   = OFF_WHT + (size_t)3 * H * H * 2;     // 32768*512 bf16
static constexpr size_t OFF_G    = OFF_XB  + (size_t)M_TOT * D * 2;     // 3 * 1024*32*512 f32

// ---------------------------------------------------------------------------
// CDNA5 async global->LDS copy (16B per lane), tracked by ASYNCcnt.
// LDS destination = low 32 bits of the generic pointer (flat->LDS mapping
// uses addr[31:0]).
// ---------------------------------------------------------------------------
__device__ __forceinline__ void async_ld_b128(void* lds, const void* g) {
  asm volatile("global_load_async_to_lds_b128 %0, %1, off"
               :: "v"((unsigned)(size_t)lds), "v"(g)
               : "memory");
}
__device__ __forceinline__ void wait_async0() {
  asm volatile("s_wait_asynccnt 0" ::: "memory");
}

// ---------------------------------------------------------------------------
// init: zero grid-barrier state, seed h ping-pong buffers with h0
// ---------------------------------------------------------------------------
__global__ __launch_bounds__(256) void init_state(const float* __restrict__ h0,
                                                  float* __restrict__ hf,
                                                  __bf16* __restrict__ hb,
                                                  unsigned* __restrict__ sync) {
  int i = blockIdx.x * 256 + threadIdx.x;
  if (i < 4) sync[i] = 0u;
  if (i < B * H) {
    float v = h0[i];
    hf[i] = v;
    hb[i] = (__bf16)v;
  }
}

// ---------------------------------------------------------------------------
// f32 -> bf16 conversion of x (vectorized, 4 elems/thread)
// ---------------------------------------------------------------------------
__global__ __launch_bounds__(256) void convert_x(const float* __restrict__ x,
                                                 __bf16* __restrict__ xb) {
  size_t i = ((size_t)blockIdx.x * 256 + threadIdx.x) * 4;
  float4 v = *(const float4*)(x + i);
  xb[i + 0] = (__bf16)v.x;
  xb[i + 1] = (__bf16)v.y;
  xb[i + 2] = (__bf16)v.z;
  xb[i + 3] = (__bf16)v.w;
}

// ---------------------------------------------------------------------------
// transpose + convert one 512x512 weight: wt[h][d] = (bf16) w[d][h]
// ---------------------------------------------------------------------------
__global__ __launch_bounds__(256) void transpose_w(const float* __restrict__ w,
                                                   __bf16* __restrict__ wt) {
  int hcol = blockIdx.x * 16 + threadIdx.x;   // coalesced read along h
  int drow = blockIdx.y * 16 + threadIdx.y;
  wt[(size_t)hcol * 512 + drow] = (__bf16)w[(size_t)drow * 512 + hcol];
}

// ---------------------------------------------------------------------------
// grid-wide barrier for the persistent scan kernel
// ---------------------------------------------------------------------------
__device__ __forceinline__ void grid_barrier(unsigned* cnt, unsigned* gen, unsigned nb) {
  __syncthreads();
  if (threadIdx.x == 0) {
    unsigned g = __hip_atomic_load(gen, __ATOMIC_ACQUIRE, __HIP_MEMORY_SCOPE_AGENT);
    __threadfence();
    unsigned a = __hip_atomic_fetch_add(cnt, 1u, __ATOMIC_ACQ_REL, __HIP_MEMORY_SCOPE_AGENT);
    if (a == nb - 1u) {
      __hip_atomic_store(cnt, 0u, __ATOMIC_RELAXED, __HIP_MEMORY_SCOPE_AGENT);
      __hip_atomic_fetch_add(gen, 1u, __ATOMIC_ACQ_REL, __HIP_MEMORY_SCOPE_AGENT);
    } else {
      while (__hip_atomic_load(gen, __ATOMIC_ACQUIRE, __HIP_MEMORY_SCOPE_AGENT) == g) {
        __builtin_amdgcn_s_sleep(2);
      }
    }
  }
  __syncthreads();
}

// ---------------------------------------------------------------------------
// Phase 1 GEMM: G[g][t][b][h] = x[b][t][:] @ Wx[g] (+ bias for z,r)
// block = 256 threads (8 waves), tile BM=128 x BN=64, K-step 32,
// double-buffered LDS filled by async global->LDS copies.
// Wave grid 4(M) x 2(N); each wave owns a 32x32 tile = 2x2 WMMA accumulators.
// ---------------------------------------------------------------------------
__global__ __launch_bounds__(256) void gemm_xproj(const __bf16* __restrict__ xb,
                                                  const __bf16* __restrict__ wxt,
                                                  const float* __restrict__ bz,
                                                  const float* __restrict__ br,
                                                  float* __restrict__ G) {
  constexpr int BM = 128, BN = 64, BK = 32, NK = D / BK;
  __shared__ __align__(16) __bf16 As[2][BM * BK];   // 2 x 8 KB
  __shared__ __align__(16) __bf16 Bs[2][BN * BK];   // 2 x 4 KB

  const int g       = blockIdx.z;
  const int rowbase = blockIdx.x * BM;
  const int colbase = blockIdx.y * BN;
  const int tid  = threadIdx.x;
  const int lane = tid & 31;
  const int wid  = tid >> 5;
  const int wm   = wid & 3;      // 0..3 -> 32-row strip
  const int wn   = wid >> 2;     // 0..1 -> 32-col strip
  const int hlf  = lane >> 4;    // 0/1 half-wave
  const int l15  = lane & 15;

  const __bf16* wt = wxt + (size_t)g * D * H;

  // bias folded into accumulator init (z and r only)
  float biasv[2] = {0.f, 0.f};
  if (g < 2) {
    const float* bp = (g == 0) ? bz : br;
    biasv[0] = bp[colbase + wn * 32 + l15];
    biasv[1] = bp[colbase + wn * 32 + 16 + l15];
  }

  v8f acc[2][2];
#pragma unroll
  for (int m = 0; m < 2; ++m)
#pragma unroll
    for (int n = 0; n < 2; ++n)
#pragma unroll
      for (int i = 0; i < 8; ++i) acc[m][n][i] = biasv[n];

  // async stage of one K-slab into LDS buffer `buf`
  auto stage = [&](int k0, int buf) {
#pragma unroll
    for (int it = 0; it < 2; ++it) {
      int idx = tid + it * 256;            // 0..511 chunks of 8 bf16
      int row = idx >> 2, q = idx & 3;
      async_ld_b128(&As[buf][row * BK + q * 8],
                    &xb[(size_t)(rowbase + row) * D + k0 + q * 8]);
    }
    {
      int row = tid >> 2, q = tid & 3;
      async_ld_b128(&Bs[buf][row * BK + q * 8],
                    &wt[(size_t)(colbase + row) * D + k0 + q * 8]);
    }
  };

  stage(0, 0);
  for (int kk = 0; kk < NK; ++kk) {
    const int cur = kk & 1;
    wait_async0();        // our chunks of buf[cur] have landed in LDS
    __syncthreads();      // everyone's chunks have landed; prev reads done
    if (kk + 1 < NK) stage((kk + 1) * BK, cur ^ 1);

    v16bf af[2], bfr[2];
#pragma unroll
    for (int m = 0; m < 2; ++m) {
      const __bf16* rp = &As[cur][(wm * 32 + m * 16 + l15) * BK];
      ((float4*)&af[m])[0] = *(const float4*)&rp[hlf * 8];
      ((float4*)&af[m])[1] = *(const float4*)&rp[16 + hlf * 8];
    }
#pragma unroll
    for (int n = 0; n < 2; ++n) {
      const __bf16* rp = &Bs[cur][(wn * 32 + n * 16 + l15) * BK];
      ((float4*)&bfr[n])[0] = *(const float4*)&rp[hlf * 8];
      ((float4*)&bfr[n])[1] = *(const float4*)&rp[16 + hlf * 8];
    }
#pragma unroll
    for (int m = 0; m < 2; ++m)
#pragma unroll
      for (int n = 0; n < 2; ++n)
        acc[m][n] = __builtin_amdgcn_wmma_f32_16x16x32_bf16(
            false, af[m], false, bfr[n], (short)0, acc[m][n], false, false);
  }

  // epilogue: scatter into G[g][t][b][h]
  float* Gg = G + (size_t)g * L * B * H;
#pragma unroll
  for (int m = 0; m < 2; ++m)
#pragma unroll
    for (int n = 0; n < 2; ++n) {
      int col = colbase + wn * 32 + n * 16 + l15;
#pragma unroll
      for (int i = 0; i < 8; ++i) {
        int R  = rowbase + wm * 32 + m * 16 + i + 8 * hlf;  // row of x = b*L + t
        int b  = R >> 10;
        int tt = R & 1023;
        Gg[((size_t)tt * B + b) * H + col] = acc[m][n][i];
      }
    }
}

// ---------------------------------------------------------------------------
// Phase 2: persistent GRU scan.
// grid = 32 blocks x 64 threads (2 waves). Block owns 16 columns of H;
// wave w owns rows [16w, 16w+16). Recurrent weights (3 gates x 16 cols x 512 K,
// bf16) live in LDS for all 1024 steps. One grid barrier per step.
// ---------------------------------------------------------------------------
__global__ __launch_bounds__(64) void gru_scan(const __bf16* __restrict__ wht,
                                               const float* __restrict__ G,
                                               const float* __restrict__ bh,
                                               float* __restrict__ hf,        // [2][B][H]
                                               __bf16* __restrict__ hb,       // [2][B][H]
                                               float* __restrict__ outp,      // [B][L][H]
                                               float* __restrict__ hfin,      // [B][H]
                                               unsigned* __restrict__ sync) {
  __shared__ __align__(16) __bf16 Wls[3 * 16 * 512];   // 48 KB

  const int tid  = threadIdx.x;
  const int lane = tid & 31;
  const int wid  = tid >> 5;          // 0/1 -> m-tile
  const int hlf  = lane >> 4;
  const int l15  = lane & 15;
  const int colbase = blockIdx.x * 16;
  const int col  = colbase + l15;
  const int mrow = wid * 16;

  // preload recurrent weights for our 16 columns (all 3 gates), async path
  for (int i = tid; i < 3 * 16 * 512 / 8; i += 64) {
    int g   = i >> 10;                 // 1024 8-elem chunks per gate
    int rem = i & 1023;
    int row = rem >> 6;                // local column 0..15
    int q   = rem & 63;
    async_ld_b128(&Wls[(g * 16 + row) * 512 + q * 8],
                  &wht[((size_t)g * H + colbase + row) * H + q * 8]);
  }
  wait_async0();
  __syncthreads();

  const float bhv = bh[col];
  const unsigned nblk = gridDim.x;
  int p = 0;

  for (int t = 0; t < L; ++t) {
    const __bf16* hcur = hb + (size_t)p * B * H;
    const float*  hfc  = hf + (size_t)p * B * H;
    const float* Gz = G + ((size_t)0 * L + t) * B * H;
    const float* Gr = G + ((size_t)1 * L + t) * B * H;
    const float* Gh = G + ((size_t)2 * L + t) * B * H;

    v8f accz, accr, acch;
#pragma unroll
    for (int i = 0; i < 8; ++i) {
      int R = mrow + i + 8 * hlf;
      accz[i] = Gz[(size_t)R * H + col];
      accr[i] = Gr[(size_t)R * H + col];
      acch[i] = 0.f;
    }

#pragma unroll 2
    for (int k0 = 0; k0 < H; k0 += 32) {
      v16bf a, wz, wr, wh;
      const __bf16* ar = hcur + (size_t)(mrow + l15) * H + k0;
      ((float4*)&a)[0] = *(const float4*)&ar[hlf * 8];
      ((float4*)&a)[1] = *(const float4*)&ar[16 + hlf * 8];

      const __bf16* rz = &Wls[(0 * 16 + l15) * 512 + k0];
      const __bf16* rr = &Wls[(1 * 16 + l15) * 512 + k0];
      const __bf16* rh = &Wls[(2 * 16 + l15) * 512 + k0];
      ((float4*)&wz)[0] = *(const float4*)&rz[hlf * 8];
      ((float4*)&wz)[1] = *(const float4*)&rz[16 + hlf * 8];
      ((float4*)&wr)[0] = *(const float4*)&rr[hlf * 8];
      ((float4*)&wr)[1] = *(const float4*)&rr[16 + hlf * 8];
      ((float4*)&wh)[0] = *(const float4*)&rh[hlf * 8];
      ((float4*)&wh)[1] = *(const float4*)&rh[16 + hlf * 8];

      accz = __builtin_amdgcn_wmma_f32_16x16x32_bf16(false, a, false, wz,
                                                     (short)0, accz, false, false);
      accr = __builtin_amdgcn_wmma_f32_16x16x32_bf16(false, a, false, wr,
                                                     (short)0, accr, false, false);
      acch = __builtin_amdgcn_wmma_f32_16x16x32_bf16(false, a, false, wh,
                                                     (short)0, acch, false, false);
    }

    // activations + state update + outputs
    float*  hf_nxt = hf + (size_t)(1 - p) * B * H;
    __bf16* hb_nxt = hb + (size_t)(1 - p) * B * H;
#pragma unroll
    for (int i = 0; i < 8; ++i) {
      int R = mrow + i + 8 * hlf;                  // batch index
      float hold = hfc[(size_t)R * H + col];
      float z  = 1.f / (1.f + __expf(-accz[i]));
      float r  = 1.f / (1.f + __expf(-accr[i]));
      float hc = tanhf(Gh[(size_t)R * H + col] + r * (acch[i] + bhv));
      float hn = z * hold + (1.f - z) * hc;
      outp[((size_t)R * L + t) * H + col] = hn;
      hf_nxt[(size_t)R * H + col] = hn;
      hb_nxt[(size_t)R * H + col] = (__bf16)hn;
      if (t == L - 1) hfin[(size_t)R * H + col] = hn;
    }

    grid_barrier(sync, sync + 1, nblk);
    p ^= 1;
  }
}

// ---------------------------------------------------------------------------
// launch
// ---------------------------------------------------------------------------
extern "C" void kernel_launch(void* const* d_in, const int* in_sizes, int n_in,
                              void* d_out, int out_size, void* d_ws, size_t ws_size,
                              hipStream_t stream) {
  const float* x   = (const float*)d_in[0];
  const float* h0  = (const float*)d_in[1];
  const float* Wxz = (const float*)d_in[2];
  const float* Whz = (const float*)d_in[3];
  const float* bz  = (const float*)d_in[4];
  const float* Wxr = (const float*)d_in[5];
  const float* Whr = (const float*)d_in[6];
  const float* br  = (const float*)d_in[7];
  const float* Wxh = (const float*)d_in[8];
  const float* Whh = (const float*)d_in[9];
  const float* bh  = (const float*)d_in[10];

  float* outp = (float*)d_out;
  char*  ws   = (char*)d_ws;

  unsigned* sync = (unsigned*)(ws + OFF_SYNC);
  float*    hf   = (float*)(ws + OFF_HF);
  __bf16*   hb   = (__bf16*)(ws + OFF_HB);
  __bf16*   wxt  = (__bf16*)(ws + OFF_WXT);
  __bf16*   wht  = (__bf16*)(ws + OFF_WHT);
  __bf16*   xb   = (__bf16*)(ws + OFF_XB);
  float*    G    = (float*)(ws + OFF_G);

  init_state<<<64, 256, 0, stream>>>(h0, hf, hb, sync);
  convert_x<<<(M_TOT * D) / (256 * 4), 256, 0, stream>>>(x, xb);

  dim3 tb(16, 16), tg(32, 32);
  transpose_w<<<tg, tb, 0, stream>>>(Wxz, wxt + 0 * (size_t)D * H);
  transpose_w<<<tg, tb, 0, stream>>>(Wxr, wxt + 1 * (size_t)D * H);
  transpose_w<<<tg, tb, 0, stream>>>(Wxh, wxt + 2 * (size_t)D * H);
  transpose_w<<<tg, tb, 0, stream>>>(Whz, wht + 0 * (size_t)H * H);
  transpose_w<<<tg, tb, 0, stream>>>(Whr, wht + 1 * (size_t)H * H);
  transpose_w<<<tg, tb, 0, stream>>>(Whh, wht + 2 * (size_t)H * H);

  gemm_xproj<<<dim3(M_TOT / 128, H / 64, 3), 256, 0, stream>>>(xb, wxt, bz, br, G);

  gru_scan<<<H / 16, 64, 0, stream>>>(wht, G, bh, hf, hb, outp,
                                      outp + (size_t)B * L * H, sync);
}
